// Multihead_SelfAttention_26980984553808
// MI455X (gfx1250) — matrix operand order repaired and checked by reading, not compiled, not run on hardware
//
#include <hip/hip_runtime.h>
#include <hip/hip_bf16.h>
#include <math.h>

// ---------------------------------------------------------------------------
// Multi-head self-attention for MI455X (gfx1250), wave32 + WMMA bf16 +
// async global->LDS staging (ASYNCcnt).
// Pipeline:
//   0) Wq,Wk,Wv,Wo -> bf16, transposed to [N][K]         (wtrans_kernel)
//   1) Qpb = bf16(q@Wq+bq) ; Kpb = bf16(k@Wk+bk)         (gemm<A_F32,CM_BF16>)
//      Vpt = bf16((v@Wv+bv)^T per head)                  (gemm<A_F32,CM_BF16T>)
//   2) attn = mask((Qpb@Kpb^T)/8)  (f32, d_out tail)     (gemm<A_BF16,CM_F32>)
//   3) softmax rows in place                             (softmax_kernel)
//   4) ctxb = bf16(attn @ Vpt^T)  (concat layout)        (gemm<A_F32,CM_BF16>)
//   5) proj = ctxb @ Wo + bo  (f32)                      (gemm<A_BF16,CM_F32>)
//   6) out = LayerNorm(proj + q)                         (ln_kernel)
// ---------------------------------------------------------------------------

typedef __attribute__((ext_vector_type(16))) __bf16 v16bf;
typedef __attribute__((ext_vector_type(8)))  float  v8f;

union FragBF { uint4 u[2]; v16bf v; };          // 32B = 16 bf16
union AccF   { v8f v; float f[8]; };
union Pack8  { unsigned short s[8]; uint4 q; };

__device__ __forceinline__ unsigned short f2bf(float f) {
    unsigned u = __float_as_uint(f);
    u += 0x7FFFu + ((u >> 16) & 1u);            // round-to-nearest-even
    return (unsigned short)(u >> 16);
}

// gfx1250 async global->LDS copy (tracked by ASYNCcnt), 16B per lane.
__device__ __forceinline__ void async_g2lds_b128(unsigned lds_off, const void* g) {
    asm volatile("global_load_async_to_lds_b128 %0, %1, off"
                 :: "v"(lds_off), "v"((unsigned long long)(uintptr_t)g)
                 : "memory");
}
__device__ __forceinline__ void wait_async0() {
    asm volatile("s_wait_asynccnt 0x0" ::: "memory");
}

// Tile geometry: 256 threads = 8 waves (wave32); block tile 64x128, K-step 32.
// Waves 2(M) x 4(N), each wave a 32x32 sub-tile = 2x2 WMMA 16x16x32 frags.
#define MT 64
#define NT 128
#define KT 32
#define AT_STRIDE 40
#define BT_STRIDE 40

#define CM_F32   0
#define CM_BF16  1
#define CM_BF16T 2

// A: [M][K]  (f32 if A_F32, else bf16) ; B: [N][K] bf16 (always "key-major")
// C: f32 [M][N] | bf16 [M][N] | bf16 transposed [N][M]  (CMODE)
template <bool A_F32, int CMODE>
__global__ __launch_bounds__(256)
void wmma_gemm(const void* __restrict__ Aptr, long lda, long sAb, long sAh,
               const void* __restrict__ Bptr, long ldb, long sBb, long sBh,
               const float* __restrict__ bias,
               void* __restrict__ Cptr, long ldc, long sCb, long sCh,
               int M, int N, int K, float alpha,
               const unsigned char* __restrict__ mask, int Hdiv)
{
    __shared__ __align__(16) unsigned short sA[MT * AT_STRIDE];
    __shared__ __align__(16) unsigned short sB[NT * BT_STRIDE];

    const int tid  = threadIdx.x;
    const int lane = tid & 31;
    const int wave = tid >> 5;
    const int wm   = wave >> 2;          // 0..1
    const int wn   = wave & 3;           // 0..3
    const int r16  = lane & 15;
    const int h    = lane >> 4;

    const int z  = blockIdx.z;
    const int bb = z / Hdiv;
    const int hh = z % Hdiv;
    const long aOff = (long)bb * sAb + (long)hh * sAh;
    const long bOff = (long)bb * sBb + (long)hh * sBh;
    const long cOff = (long)bb * sCb + (long)hh * sCh;

    const float*          Af  = (const float*)Aptr + aOff;          // A_F32
    const unsigned short* A16 = (const unsigned short*)Aptr + aOff; // !A_F32
    const unsigned short* B16 = (const unsigned short*)Bptr + bOff;
    const unsigned char*  maskB =
        mask ? (mask + (long)bb * (long)M * (long)N) : nullptr;

    const int blockM = blockIdx.y * MT;
    const int blockN = blockIdx.x * NT;

    v8f acc[2][2];
    const v8f vzero = {0.f, 0.f, 0.f, 0.f, 0.f, 0.f, 0.f, 0.f};
    acc[0][0] = vzero; acc[0][1] = vzero; acc[1][0] = vzero; acc[1][1] = vzero;
    const uint4 z4 = {0u, 0u, 0u, 0u};

    // per-thread staging coordinates
    const int ar  = tid >> 2;            // A: 0..63 rows
    const int acg = (tid & 3) * 8;       // A: col group (8 elems)
    const int bn  = tid >> 1;            // B: 0..127 rows
    const int bcg = (tid & 1) * 16;      // B: col group (16 elems)

    for (int k0 = 0; k0 < K; k0 += KT) {
        // ---------------- stage A tile (MT x KT bf16 in LDS) ----------------
        {
            const long gm = (long)blockM + ar;
            if (A_F32) {
                Pack8 pk;
                if (gm < M) {
                    const float* src = Af + gm * lda + k0 + acg;
#pragma unroll
                    for (int i = 0; i < 8; ++i) pk.s[i] = f2bf(src[i]);
                    if (k0 + KT < K) __builtin_prefetch(src + KT, 0, 0);
                } else {
#pragma unroll
                    for (int i = 0; i < 8; ++i) pk.s[i] = 0;
                }
                *reinterpret_cast<uint4*>(&sA[ar * AT_STRIDE + acg]) = pk.q;
            } else {
                const unsigned ldsA =
                    (unsigned)(uintptr_t)&sA[ar * AT_STRIDE + acg];
                if (gm < M) {
                    const unsigned short* src = A16 + gm * lda + k0 + acg;
                    async_g2lds_b128(ldsA, src);
                    if (k0 + KT < K) __builtin_prefetch(src + KT, 0, 0);
                } else {
                    *reinterpret_cast<uint4*>(&sA[ar * AT_STRIDE + acg]) = z4;
                }
            }
        }
        // ------------- stage B tile (NT x KT bf16, straight copy) -----------
        {
            const long gn = (long)blockN + bn;
            if (gn < N) {
                const unsigned short* src = B16 + gn * ldb + k0 + bcg;
                async_g2lds_b128(
                    (unsigned)(uintptr_t)&sB[bn * BT_STRIDE + bcg], src);
                async_g2lds_b128(
                    (unsigned)(uintptr_t)&sB[bn * BT_STRIDE + bcg + 8], src + 8);
                if (k0 + KT < K) __builtin_prefetch(src + KT, 0, 0);
            } else {
                *reinterpret_cast<uint4*>(&sB[bn * BT_STRIDE + bcg])     = z4;
                *reinterpret_cast<uint4*>(&sB[bn * BT_STRIDE + bcg + 8]) = z4;
            }
        }
        wait_async0();                       // ASYNCcnt -> 0 (LDS writes done)
        __syncthreads();

        // ---- fragments per gfx1250 16-bit WMMA layouts ----
        FragBF fa[2], fb[2];
#pragma unroll
        for (int i = 0; i < 2; ++i) {
            const int row = wm * 32 + i * 16 + r16;
            fa[i].u[0] = *reinterpret_cast<const uint4*>(
                &sA[row * AT_STRIDE + h * 8]);          // K = h*8 + 0..7
            fa[i].u[1] = *reinterpret_cast<const uint4*>(
                &sA[row * AT_STRIDE + 16 + h * 8]);     // K = 16 + h*8 + 0..7
        }
#pragma unroll
        for (int j = 0; j < 2; ++j) {
            const int col = wn * 32 + j * 16 + r16;
            fb[j].u[0] = *reinterpret_cast<const uint4*>(
                &sB[col * BT_STRIDE + h * 8]);          // K = h*8 + 0..7
            fb[j].u[1] = *reinterpret_cast<const uint4*>(
                &sB[col * BT_STRIDE + 16 + h * 8]);     // K = 16 + h*8 + 0..7
        }
#pragma unroll
        for (int i = 0; i < 2; ++i)
#pragma unroll
            for (int j = 0; j < 2; ++j)
                acc[i][j] = __builtin_amdgcn_wmma_f32_16x16x32_bf16(
                    false, fa[i].v, false, fb[j].v,
                    (short)0, acc[i][j], false, false);
        __syncthreads();
    }

    // -------------------------- epilogue -----------------------------------
#pragma unroll
    for (int i = 0; i < 2; ++i) {
#pragma unroll
        for (int j = 0; j < 2; ++j) {
            AccF c; c.v = acc[i][j];
            const int n_g = blockN + wn * 32 + j * 16 + r16;
            if (CMODE == CM_BF16T) {
                // transposed bf16: lane's 8 elements are contiguous in M
                Pack8 pk;
                const float bia = bias ? bias[n_g < N ? n_g : 0] : 0.f;
#pragma unroll
                for (int e = 0; e < 8; ++e)
                    pk.s[e] = f2bf(c.f[e] * alpha + bia);
                const int m0 = blockM + wm * 32 + i * 16 + 8 * h;
                if (n_g < N && m0 + 7 < M) {
                    unsigned short* Ct = (unsigned short*)Cptr + cOff;
                    *reinterpret_cast<uint4*>(&Ct[(long)n_g * ldc + m0]) = pk.q;
                }
            } else {
#pragma unroll
                for (int e = 0; e < 8; ++e) {
                    const int m_g = blockM + wm * 32 + i * 16 + e + 8 * h;
                    if (m_g < M && n_g < N) {
                        float val = c.f[e] * alpha;
                        if (bias) val += bias[n_g];
                        if (CMODE == CM_F32) {
                            if (maskB && maskB[(long)m_g * N + n_g]) val = -1e9f;
                            ((float*)Cptr + cOff)[(long)m_g * ldc + n_g] = val;
                        } else {
                            ((unsigned short*)Cptr + cOff)[(long)m_g * ldc + n_g]
                                = f2bf(val);
                        }
                    }
                }
            }
        }
    }
}

// ---------------- weight transpose + f32->bf16 (one-off pass) --------------
// W [K][N] f32  ->  Wt [N][K] bf16 ; K,N multiples of 32 (1024 here).
__global__ __launch_bounds__(256)
void wtrans_kernel(const float* __restrict__ W, unsigned short* __restrict__ Wt,
                   int K, int N)
{
    __shared__ float t[32][33];
    const int tx = threadIdx.x & 31, ty = threadIdx.x >> 5;  // 32 x 8
    const int k0 = blockIdx.y * 32, n0 = blockIdx.x * 32;
#pragma unroll
    for (int i = 0; i < 4; ++i)
        t[ty + i * 8][tx] = W[(long)(k0 + ty + i * 8) * N + n0 + tx];
    __syncthreads();
#pragma unroll
    for (int i = 0; i < 4; ++i)
        Wt[(long)(n0 + ty + i * 8) * K + k0 + tx] = f2bf(t[tx][ty + i * 8]);
}

// ------------------------- row softmax (wave32) ----------------------------
__device__ __forceinline__ float wave_max(float v) {
#pragma unroll
    for (int o = 16; o > 0; o >>= 1) v = fmaxf(v, __shfl_xor(v, o, 32));
    return v;
}
__device__ __forceinline__ float wave_sum(float v) {
#pragma unroll
    for (int o = 16; o > 0; o >>= 1) v += __shfl_xor(v, o, 32);
    return v;
}

__global__ __launch_bounds__(256)
void softmax_kernel(float* __restrict__ attn, int Scols)
{
    __shared__ float red[8];
    const long row = blockIdx.x;
    float* p = attn + row * (long)Scols;
    const int tid = threadIdx.x;
    const int lane = tid & 31, wave = tid >> 5;

    float x[4];
    float m = -3.4e38f;
#pragma unroll
    for (int i = 0; i < 4; ++i) { x[i] = p[tid + i * 256]; m = fmaxf(m, x[i]); }
    m = wave_max(m);
    if (lane == 0) red[wave] = m;
    __syncthreads();
    m = red[lane & 7];
#pragma unroll
    for (int o = 4; o > 0; o >>= 1) m = fmaxf(m, __shfl_xor(m, o, 32));

    float s = 0.f;
#pragma unroll
    for (int i = 0; i < 4; ++i) { x[i] = __expf(x[i] - m); s += x[i]; }
    s = wave_sum(s);
    __syncthreads();
    if (lane == 0) red[wave] = s;
    __syncthreads();
    s = red[lane & 7];
#pragma unroll
    for (int o = 4; o > 0; o >>= 1) s += __shfl_xor(s, o, 32);
    const float inv = 1.0f / s;
#pragma unroll
    for (int i = 0; i < 4; ++i) p[tid + i * 256] = x[i] * inv;
}

// ------------------- residual + LayerNorm (weight=1, bias=0) ---------------
__global__ __launch_bounds__(256)
void ln_kernel(const float* __restrict__ proj, const float* __restrict__ resid,
               float* __restrict__ out, int Dcols)
{
    __shared__ float red[16];
    const long row = blockIdx.x;
    const float* pp = proj  + row * (long)Dcols;
    const float* rr = resid + row * (long)Dcols;
    float* oo = out + row * (long)Dcols;
    const int tid = threadIdx.x;
    const int lane = tid & 31, wave = tid >> 5;

    float x[4], sum = 0.f, sq = 0.f;
#pragma unroll
    for (int i = 0; i < 4; ++i) {
        x[i] = pp[tid + i * 256] + rr[tid + i * 256];
        sum += x[i]; sq += x[i] * x[i];
    }
    sum = wave_sum(sum); sq = wave_sum(sq);
    if (lane == 0) { red[wave] = sum; red[wave + 8] = sq; }
    __syncthreads();
    sum = red[lane & 7]; sq = red[8 + (lane & 7)];
#pragma unroll
    for (int o = 4; o > 0; o >>= 1) { sum += __shfl_xor(sum, o, 32); sq += __shfl_xor(sq, o, 32); }
    const float invD = 1.0f / (float)Dcols;
    const float mean = sum * invD;
    float var = fmaxf(sq * invD - mean * mean, 0.f);
    const float rstd = rsqrtf(var + 1e-5f);
#pragma unroll
    for (int i = 0; i < 4; ++i) oo[tid + i * 256] = (x[i] - mean) * rstd;
}

// ---------------------------------------------------------------------------
extern "C" void kernel_launch(void* const* d_in, const int* in_sizes, int n_in,
                              void* d_out, int out_size, void* d_ws, size_t ws_size,
                              hipStream_t stream)
{
    constexpr int  Bn = 4, S = 1024, D = 1024, H = 16, DK = 64;
    constexpr long M4 = (long)Bn * S;            // 4096
    constexpr long MM = 1024l * 1024l;
    (void)in_sizes; (void)n_in; (void)out_size; (void)ws_size;

    const float* q    = (const float*)d_in[0];
    const float* k    = (const float*)d_in[1];
    const float* v    = (const float*)d_in[2];
    const unsigned char* mask = (const unsigned char*)d_in[3];
    const float* Wq = (const float*)d_in[4];
    const float* bq = (const float*)d_in[5];
    const float* Wk = (const float*)d_in[6];
    const float* bk = (const float*)d_in[7];
    const float* Wv = (const float*)d_in[8];
    const float* bv = (const float*)d_in[9];
    const float* Wo = (const float*)d_in[10];
    const float* bo = (const float*)d_in[11];

    // workspace (bf16 stored as ushort)
    unsigned short* Qpb  = (unsigned short*)d_ws;   // [4096,1024]
    unsigned short* Kpb  = Qpb  + 4 * MM;           // [4096,1024]
    unsigned short* Vpt  = Kpb  + 4 * MM;           // [B][H][64][S]
    unsigned short* ctxb = Vpt  + 4 * MM;           // [4096, H*DV]
    unsigned short* Wqt  = ctxb + 4 * MM;           // [1024,1024] transposed
    unsigned short* Wkt  = Wqt  + MM;
    unsigned short* Wvt  = Wkt  + MM;
    unsigned short* Wot  = Wvt  + MM;
    float*          proj = (float*)(Wot + MM);      // [4096,1024] f32

    float* out  = (float*)d_out;                    // [B,S,D]
    float* attn = out + (long)Bn * S * D;           // [B,H,S,S]

    const dim3 blk(256);

    // 0) transpose+convert weights to bf16 [N][K]
    {
        dim3 g(D / 32, D / 32, 1);
        wtrans_kernel<<<g, blk, 0, stream>>>(Wq, Wqt, D, H * DK);
        wtrans_kernel<<<g, blk, 0, stream>>>(Wk, Wkt, D, H * DK);
        wtrans_kernel<<<g, blk, 0, stream>>>(Wv, Wvt, D, H * DK);
        wtrans_kernel<<<g, blk, 0, stream>>>(Wo, Wot, H * DK, D);
    }

    // 1) Projections (A f32, B bf16-transposed)
    {
        dim3 g(D / NT, M4 / MT, 1);
        wmma_gemm<true, CM_BF16><<<g, blk, 0, stream>>>(
            q, D, 0, 0, Wqt, D, 0, 0, bq,
            Qpb, H * DK, 0, 0, (int)M4, H * DK, D, 1.0f, nullptr, 1);
        wmma_gemm<true, CM_BF16><<<g, blk, 0, stream>>>(
            k, D, 0, 0, Wkt, D, 0, 0, bk,
            Kpb, H * DK, 0, 0, (int)M4, H * DK, D, 1.0f, nullptr, 1);
    }
    {   // V projection, batched over b, per-head-transposed output
        dim3 g(D / NT, S / MT, Bn);
        wmma_gemm<true, CM_BF16T><<<g, blk, 0, stream>>>(
            v, D, (long)S * D, 0, Wvt, D, 0, 0, bv,
            Vpt, S, (long)H * DK * S, 0, S, H * DK, D, 1.0f, nullptr, 1);
    }

    // 2) Scores: per (b,h) [S,dk] x [S,dk]^T, scaled + masked -> attn (f32)
    {
        dim3 g(S / NT, S / MT, Bn * H);
        wmma_gemm<false, CM_F32><<<g, blk, 0, stream>>>(
            Qpb, H * DK, (long)S * H * DK, DK,
            Kpb, H * DK, (long)S * H * DK, DK,
            nullptr,
            attn, S, (long)H * S * S, (long)S * S,
            S, S, DK, 0.125f, mask, H);
    }

    // 3) Softmax rows in place
    softmax_kernel<<<Bn * H * S, blk, 0, stream>>>(attn, S);

    // 4) Context: per (b,h) [S,S](f32) x Vpt[dv][S]^T -> ctxb (bf16 concat)
    {
        dim3 g((DK + NT - 1) / NT, S / MT, Bn * H);
        wmma_gemm<true, CM_BF16><<<g, blk, 0, stream>>>(
            attn, S, (long)H * S * S, (long)S * S,
            Vpt, S, (long)H * DK * S, (long)DK * S,
            nullptr,
            ctxb, H * DK, (long)S * H * DK, DK,
            S, DK, S, 1.0f, nullptr, H);
    }

    // 5) Output projection: ctxb(bf16) x Wot(bf16) + bo -> proj (f32)
    {
        dim3 g(D / NT, M4 / MT, 1);
        wmma_gemm<false, CM_F32><<<g, blk, 0, stream>>>(
            ctxb, H * DK, 0, 0, Wot, H * DK, 0, 0, bo,
            proj, D, 0, 0, (int)M4, D, H * DK, 1.0f, nullptr, 1);
    }

    // 6) Residual + LayerNorm
    ln_kernel<<<(int)M4, blk, 0, stream>>>(proj, q, out, D);
}
